// PointTransformerAttention_62878321214016
// MI455X (gfx1250) — matrix operand order repaired and checked by reading, not compiled, not run on hardware
//
#include <hip/hip_runtime.h>

// ---------------------------------------------------------------------------
// PointTransformer attention for MI455X (gfx1250), wave32.
// B=4, S=1024, N=16384, D=128, KNN_K=16.
//
// Pipeline (all on `stream`, dependencies via stream order):
//   1. cvt:     k_feat,Wv  -> bf16 buffers (for WMMA)
//   2. prep:    column sums of Wq/Wk/W2, scalar sums of bq/bk/b2
//   3. rowsum:  sumk[b,n] = k_feat . colsum(Wk) + sum(bk)
//   4. gemm_v:  v = k_feat @ Wv^T + bv  via v_wmma_f32_16x16x32_bf16
//   5. knn:     top-16 neighbors per query (wave per query)
//   6. scores:  score = sumq - sumk + sum(b2) + cs2 . relu(W1.rel + b1),
//               softmax over 16 -> attn
//   7. colsum:  deterministic sum of attn over S per (b, j)
//   8. out:     out = sum_j attn_norm_j * v[idx_j]
// ---------------------------------------------------------------------------

#define B_  4
#define S_  1024
#define N_  16384
#define D_  128
#define KK_ 16
#define EPS_ 1e-6f

typedef unsigned int   u32;
typedef unsigned short u16;
typedef __attribute__((ext_vector_type(16))) __bf16 bfx16;
typedef __attribute__((ext_vector_type(8)))  float  fx8;

struct alignas(8) US4 { u16 x, y, z, w; };

__device__ __forceinline__ u16 f2bf(float f) {
    u32 u = __float_as_uint(f);
    u += 0x7fffu + ((u >> 16) & 1u);   // round-to-nearest-even (finite inputs)
    return (u16)(u >> 16);
}

__device__ __forceinline__ float wave_sum32(float v) {
    #pragma unroll
    for (int off = 16; off; off >>= 1) v += __shfl_xor(v, off, 32);
    return v;
}

// ---------------------------------------------------------------------------
// 1. Convert k_feat (B*N*D) and Wv (D*D) fp32 -> bf16
// ---------------------------------------------------------------------------
__global__ __launch_bounds__(256) void pt_cvt_kernel(
    const float* __restrict__ kf, const float* __restrict__ wv,
    u16* __restrict__ kb, u16* __restrict__ wvb)
{
    const size_t KQ = (size_t)B_ * N_ * D_ / 4;   // float4 count for k_feat
    size_t t = (size_t)blockIdx.x * 256 + threadIdx.x;
    if (t < KQ) {
        float4 f = reinterpret_cast<const float4*>(kf)[t];
        US4 r = { f2bf(f.x), f2bf(f.y), f2bf(f.z), f2bf(f.w) };
        reinterpret_cast<US4*>(kb)[t] = r;
    } else if (t < KQ + (D_ * D_ / 4)) {
        size_t j = t - KQ;
        float4 f = reinterpret_cast<const float4*>(wv)[j];
        US4 r = { f2bf(f.x), f2bf(f.y), f2bf(f.z), f2bf(f.w) };
        reinterpret_cast<US4*>(wvb)[j] = r;
    }
}

// ---------------------------------------------------------------------------
// 2. prep: prep[0..127]=colsum(Wq), [128..255]=colsum(Wk), [256..383]=colsum(W2)
//          prep[384]=sum(bq), [385]=sum(bk), [386]=sum(b2)
// ---------------------------------------------------------------------------
__global__ __launch_bounds__(128) void pt_prep_kernel(
    const float* __restrict__ Wq, const float* __restrict__ Wk,
    const float* __restrict__ W2, const float* __restrict__ bq,
    const float* __restrict__ bk, const float* __restrict__ b2,
    float* __restrict__ prep)
{
    int i = threadIdx.x;                  // 0..127
    float a = 0.f, c = 0.f, e = 0.f;
    for (int d = 0; d < D_; ++d) {
        a += Wq[d * D_ + i];
        c += Wk[d * D_ + i];
        e += W2[d * D_ + i];
    }
    prep[i] = a; prep[128 + i] = c; prep[256 + i] = e;
    __shared__ float r1[128], r2[128], r3[128];
    r1[i] = bq[i]; r2[i] = bk[i]; r3[i] = b2[i];
    __syncthreads();
    for (int st = 64; st; st >>= 1) {
        if (i < st) { r1[i] += r1[i + st]; r2[i] += r2[i + st]; r3[i] += r3[i + st]; }
        __syncthreads();
    }
    if (i == 0) { prep[384] = r1[0]; prep[385] = r2[0]; prep[386] = r3[0]; }
}

// ---------------------------------------------------------------------------
// 3. sumk[row] = k_feat[row,:] . colsum(Wk) + sum(bk)   (wave per row)
// ---------------------------------------------------------------------------
__global__ __launch_bounds__(256) void pt_rowsum_kernel(
    const float* __restrict__ kf, const float* __restrict__ prep,
    float* __restrict__ sumk)
{
    const int wave = threadIdx.x >> 5, lane = threadIdx.x & 31;
    const int row = blockIdx.x * 8 + wave;          // 0..B*N-1
    float4 kv = *reinterpret_cast<const float4*>(kf + (size_t)row * D_ + lane * 4);
    float4 wk = *reinterpret_cast<const float4*>(prep + 128 + lane * 4);
    float part = kv.x * wk.x + kv.y * wk.y + kv.z * wk.z + kv.w * wk.w;
    float s = wave_sum32(part);
    if (lane == 0) sumk[row] = s + prep[385];
}

// ---------------------------------------------------------------------------
// 4. v = k_feat @ Wv^T + bv  — WMMA GEMM (bf16 in, f32 acc).
//    Wave computes a 16-row strip x all 128 cols. sched_group_barrier
//    template forces [16 ds_load_b128][8 v_wmma] per K-chunk so the waitcnt
//    pass emits graduated s_wait_dscnt and WMMAs overlap in-flight LDS loads
//    (instead of a full DS drain before every WMMA).
// ---------------------------------------------------------------------------
union ABfrag { uint4 q[2]; bfx16 v; };

#if defined(__has_builtin)
#  if __has_builtin(__builtin_amdgcn_sched_group_barrier)
#    define PT_SGB(mask, size) __builtin_amdgcn_sched_group_barrier((mask), (size), 0)
#  endif
#endif
#ifndef PT_SGB
#  define PT_SGB(mask, size)
#endif

__global__ __launch_bounds__(256) void pt_gemm_v_kernel(
    const u16* __restrict__ kb, const u16* __restrict__ wvb,
    const float* __restrict__ bv, float* __restrict__ vout)
{
    __shared__ alignas(16) u16 lw[D_ * D_];   // 32 KB of 320 KB WGP LDS
    {   // cooperative stage of Wv (bf16)
        const uint4* src = reinterpret_cast<const uint4*>(wvb);
        uint4* dst = reinterpret_cast<uint4*>(lw);
        for (int i = threadIdx.x; i < (D_ * D_ * 2) / 16; i += 256) dst[i] = src[i];
    }
    __syncthreads();

    const int wave = threadIdx.x >> 5;
    const int lane = threadIdx.x & 31;
    const int half = lane >> 4;       // lane half selects K sub-range (ISA 7.12.2)
    const int l16  = lane & 15;
    const int m0   = (blockIdx.x * 8 + wave) * 16;   // row tile base
    const int rowA = m0 + l16;

    // A fragments: 16x32 bf16 per chunk; lanes 0-15 hold K = {0..7,16..23},
    // lanes 16-31 hold K = {8..15,24..31} of row M=lane%16.
    ABfrag a[4];
    const u16* arow = kb + (size_t)rowA * D_;
    #pragma unroll
    for (int c = 0; c < 4; ++c) {
        int K0 = c * 32;
        a[c].q[0] = *reinterpret_cast<const uint4*>(arow + K0 + half * 8);
        a[c].q[1] = *reinterpret_cast<const uint4*>(arow + K0 + 16 + half * 8);
    }

    fx8 acc[8];
    #pragma unroll
    for (int nt = 0; nt < 8; ++nt)
        acc[nt] = (fx8){0.f, 0.f, 0.f, 0.f, 0.f, 0.f, 0.f, 0.f};

    #pragma unroll
    for (int c = 0; c < 4; ++c) {
        // preload all 8 B fragments for this K-chunk (independent ds_loads)
        ABfrag bf[8];
        #pragma unroll
        for (int nt = 0; nt < 8; ++nt) {
            const int col = nt * 16 + l16;
            const u16* p = &lw[col * D_ + c * 32 + half * 8];
            bf[nt].q[0] = *reinterpret_cast<const uint4*>(p);
            bf[nt].q[1] = *reinterpret_cast<const uint4*>(p + 16);
        }
        #pragma unroll
        for (int nt = 0; nt < 8; ++nt) {
            acc[nt] = __builtin_amdgcn_wmma_f32_16x16x32_bf16(
                false, a[c].v, false, bf[nt].v, (short)0, acc[nt], false, false);
        }
    }

    // Scheduling template for the (fully unrolled) region above:
    // per K-chunk: 16 DS reads (0x100), then 8 WMMA (0x008).
    PT_SGB(0x100, 16); PT_SGB(0x008, 8);
    PT_SGB(0x100, 16); PT_SGB(0x008, 8);
    PT_SGB(0x100, 16); PT_SGB(0x008, 8);
    PT_SGB(0x100, 16); PT_SGB(0x008, 8);

    const int rowD = m0 + half * 8;   // C/D: VGPR r holds M=r (lanes 0-15) / M=8+r
    #pragma unroll
    for (int nt = 0; nt < 8; ++nt) {
        const int col = nt * 16 + l16;
        const float bias = bv[col];
        #pragma unroll
        for (int r = 0; r < 8; ++r)
            vout[(size_t)(rowD + r) * D_ + col] = acc[nt][r] + bias;
    }
}

// ---------------------------------------------------------------------------
// 5. KNN: wave per query; per-lane register top-16 over 512 points, then
//    16-round cross-lane merge (stable: ties -> smaller global index).
// ---------------------------------------------------------------------------
__global__ __launch_bounds__(256) void pt_knn_kernel(
    const float* __restrict__ q_pos, const float* __restrict__ k_pos,
    int* __restrict__ idx)
{
    const int wave = threadIdx.x >> 5, lane = threadIdx.x & 31;
    const int q = blockIdx.x * 8 + wave;        // 0..B*S-1
    const int b = q >> 10;
    const float qx = q_pos[q * 3 + 0], qy = q_pos[q * 3 + 1], qz = q_pos[q * 3 + 2];

    float bd[KK_]; int bi[KK_];
    #pragma unroll
    for (int j = 0; j < KK_; ++j) { bd[j] = 3.4e38f; bi[j] = 0x7fffffff; }
    float worst = 3.4e38f;

    const float* kp = k_pos + (size_t)b * N_ * 3;
    for (int p = lane; p < N_; p += 32) {
        float dx = qx - kp[p * 3 + 0];
        float dy = qy - kp[p * 3 + 1];
        float dz = qz - kp[p * 3 + 2];
        float d = dx * dx + dy * dy + dz * dz;
        if (d < worst) {
            int am = 0; float mv = bd[0];
            #pragma unroll
            for (int j = 1; j < KK_; ++j) if (bd[j] > mv) { mv = bd[j]; am = j; }
            #pragma unroll
            for (int j = 0; j < KK_; ++j) if (j == am) { bd[j] = d; bi[j] = p; }
            worst = bd[0];
            #pragma unroll
            for (int j = 1; j < KK_; ++j) worst = fmaxf(worst, bd[j]);
        }
    }

    // merge: 16 rounds of (local min over unused) + wave butterfly min
    u32 used = 0;
    for (int r = 0; r < KK_; ++r) {
        float lm = 3.4e38f; int lj = 0;
        #pragma unroll
        for (int j = 0; j < KK_; ++j) {
            bool free_ = ((used >> j) & 1u) == 0u;
            if (free_ && bd[j] < lm) { lm = bd[j]; lj = j; }
        }
        int lgi = 0x7fffffff;
        #pragma unroll
        for (int j = 0; j < KK_; ++j) if (j == lj) lgi = bi[j];

        float bdv = lm; int bgi = lgi; int blane = lane;
        #pragma unroll
        for (int off = 16; off; off >>= 1) {
            float od = __shfl_xor(bdv, off, 32);
            int   og = __shfl_xor(bgi, off, 32);
            int   ol = __shfl_xor(blane, off, 32);
            if (od < bdv || (od == bdv && og < bgi)) { bdv = od; bgi = og; blane = ol; }
        }
        if (blane == lane) used |= (1u << lj);
        if (lane == 0) idx[(size_t)q * KK_ + r] = bgi;
    }
}

// ---------------------------------------------------------------------------
// 6. scores + softmax over 16 neighbors (wave per query)
// ---------------------------------------------------------------------------
__global__ __launch_bounds__(256) void pt_scores_kernel(
    const float* __restrict__ qf, const float* __restrict__ q_pos,
    const float* __restrict__ k_pos, const int* __restrict__ idx,
    const float* __restrict__ sumk, const float* __restrict__ prep,
    const float* __restrict__ W1, const float* __restrict__ b1,
    float* __restrict__ attn)
{
    const int wave = threadIdx.x >> 5, lane = threadIdx.x & 31;
    const int q = blockIdx.x * 8 + wave;
    const int b = q >> 10;

    float4 qv = *reinterpret_cast<const float4*>(qf + (size_t)q * D_ + lane * 4);
    float4 wq = *reinterpret_cast<const float4*>(prep + lane * 4);
    float sq = wave_sum32(qv.x * wq.x + qv.y * wq.y + qv.z * wq.z + qv.w * wq.w)
             + prep[384];
    const float sb2 = prep[386];
    const float qx = q_pos[q * 3 + 0], qy = q_pos[q * 3 + 1], qz = q_pos[q * 3 + 2];
    const float* cs2 = prep + 256;

    float myscore = -3.4e38f;
    for (int j = 0; j < KK_; ++j) {
        int gi = idx[(size_t)q * KK_ + j];
        const float* kpp = k_pos + ((size_t)b * N_ + gi) * 3;
        float rx = qx - kpp[0], ry = qy - kpp[1], rz = qz - kpp[2];
        float acc = 0.f;
        #pragma unroll
        for (int t = 0; t < 4; ++t) {
            int h = lane + t * 32;
            float hh = W1[h * 3 + 0] * rx + W1[h * 3 + 1] * ry + W1[h * 3 + 2] * rz + b1[h];
            acc += cs2[h] * fmaxf(hh, 0.f);
        }
        float pes = wave_sum32(acc);
        float sc = sq - sumk[(size_t)b * N_ + gi] + sb2 + pes;
        if (lane == j) myscore = sc;
    }
    // softmax within lanes 0..15 (width-16 butterflies; high half inert)
    float m = myscore;
    #pragma unroll
    for (int off = 8; off; off >>= 1) m = fmaxf(m, __shfl_xor(m, off, 16));
    float e = __expf(myscore - m);
    float s = e;
    #pragma unroll
    for (int off = 8; off; off >>= 1) s += __shfl_xor(s, off, 16);
    if (lane < KK_) attn[(size_t)q * KK_ + lane] = e / s;
}

// ---------------------------------------------------------------------------
// 7. colsum[b,j] = sum_S attn[b,s,j]   (deterministic block reduction)
// ---------------------------------------------------------------------------
__global__ __launch_bounds__(256) void pt_colsum_kernel(
    const float* __restrict__ attn, float* __restrict__ colsum)
{
    __shared__ float red[256];
    const int bj = blockIdx.x;                    // b*16 + j
    const int b = bj >> 4, j = bj & 15;
    float acc = 0.f;
    for (int s = threadIdx.x; s < S_; s += 256)
        acc += attn[((size_t)b * S_ + s) * KK_ + j];
    red[threadIdx.x] = acc; __syncthreads();
    for (int st = 128; st; st >>= 1) {
        if (threadIdx.x < st) red[threadIdx.x] += red[threadIdx.x + st];
        __syncthreads();
    }
    if (threadIdx.x == 0) colsum[bj] = red[0];
}

// ---------------------------------------------------------------------------
// 8. out[b,s,:] = sum_j attn[b,s,j]/(colsum[b,j]+eps) * v[b,idx_j,:]
// ---------------------------------------------------------------------------
__global__ __launch_bounds__(256) void pt_out_kernel(
    const float* __restrict__ attn, const float* __restrict__ colsum,
    const int* __restrict__ idx, const float* __restrict__ v,
    float* __restrict__ out)
{
    const int wave = threadIdx.x >> 5, lane = threadIdx.x & 31;
    const int q = blockIdx.x * 8 + wave;
    const int b = q >> 10;
    float4 acc = {0.f, 0.f, 0.f, 0.f};
    for (int j = 0; j < KK_; ++j) {
        int gi = idx[(size_t)q * KK_ + j];
        float w = attn[(size_t)q * KK_ + j] / (colsum[b * KK_ + j] + EPS_);
        float4 vv = *reinterpret_cast<const float4*>(
            v + ((size_t)b * N_ + gi) * D_ + lane * 4);
        acc.x += w * vv.x; acc.y += w * vv.y; acc.z += w * vv.z; acc.w += w * vv.w;
    }
    *reinterpret_cast<float4*>(out + (size_t)q * D_ + lane * 4) = acc;
}

// ---------------------------------------------------------------------------
// launch
// ---------------------------------------------------------------------------
extern "C" void kernel_launch(void* const* d_in, const int* in_sizes, int n_in,
                              void* d_out, int out_size, void* d_ws, size_t ws_size,
                              hipStream_t stream)
{
    const float* q_feat = (const float*)d_in[0];
    const float* k_feat = (const float*)d_in[1];
    const float* q_pos  = (const float*)d_in[2];
    const float* k_pos  = (const float*)d_in[3];
    const float* Wq     = (const float*)d_in[4];
    const float* bq     = (const float*)d_in[5];
    const float* Wk     = (const float*)d_in[6];
    const float* bk     = (const float*)d_in[7];
    const float* Wv     = (const float*)d_in[8];
    const float* bv     = (const float*)d_in[9];
    const float* W1     = (const float*)d_in[10];
    const float* b1     = (const float*)d_in[11];
    const float* W2     = (const float*)d_in[12];
    const float* b2     = (const float*)d_in[13];
    float* out = (float*)d_out;

    // workspace layout (byte offsets, all 256B-aligned)
    char* w = (char*)d_ws;
    const size_t KB_BYTES   = (size_t)B_ * N_ * D_ * 2;        // 16.78 MB bf16 k_feat
    const size_t WVB_BYTES  = (size_t)D_ * D_ * 2;             // 32 KB  bf16 Wv
    const size_t V_BYTES    = (size_t)B_ * N_ * D_ * 4;        // 33.55 MB f32 v
    const size_t IDX_BYTES  = (size_t)B_ * S_ * KK_ * 4;
    const size_t ATT_BYTES  = (size_t)B_ * S_ * KK_ * 4;
    const size_t SUMK_BYTES = (size_t)B_ * N_ * 4;

    size_t off = 0;
    u16*   kb     = (u16*)(w + off);   off += KB_BYTES;
    u16*   wvb    = (u16*)(w + off);   off += WVB_BYTES;
    float* v      = (float*)(w + off); off += V_BYTES;
    int*   idx    = (int*)(w + off);   off += IDX_BYTES;
    float* attn   = (float*)(w + off); off += ATT_BYTES;
    float* sumk   = (float*)(w + off); off += SUMK_BYTES;
    float* prep   = (float*)(w + off); off += 2048;
    float* colsum = (float*)(w + off); off += 256;

    const int cvt_blocks = (int)(((size_t)B_ * N_ * D_ / 4 + D_ * D_ / 4 + 255) / 256);

    pt_cvt_kernel<<<cvt_blocks, 256, 0, stream>>>(k_feat, Wv, kb, wvb);
    pt_prep_kernel<<<1, 128, 0, stream>>>(Wq, Wk, W2, bq, bk, b2, prep);
    pt_rowsum_kernel<<<(B_ * N_) / 8, 256, 0, stream>>>(k_feat, prep, sumk);
    pt_gemm_v_kernel<<<(B_ * N_ / 16) / 8, 256, 0, stream>>>(kb, wvb, bv, v);
    pt_knn_kernel<<<(B_ * S_) / 8, 256, 0, stream>>>(q_pos, k_pos, idx);
    pt_scores_kernel<<<(B_ * S_) / 8, 256, 0, stream>>>(q_feat, q_pos, k_pos, idx,
                                                        sumk, prep, W1, b1, attn);
    pt_colsum_kernel<<<B_ * KK_, 256, 0, stream>>>(attn, colsum);
    pt_out_kernel<<<(B_ * S_) / 8, 256, 0, stream>>>(attn, colsum, idx, v, out);
}
